// LSTM_52802327937084
// MI455X (gfx1250) — matrix operand order repaired and checked by reading, not compiled, not run on hardware
//
#include <hip/hip_runtime.h>
#include <hip/hip_bf16.h>

// ---------- problem constants (match reference) ----------
constexpr int kT = 256;   // time steps
constexpr int kB = 64;    // batch
constexpr int kH = 1024;  // hidden
constexpr int kL = 2;     // layers
constexpr int kG4H = 4 * kH;

typedef __attribute__((ext_vector_type(16))) __bf16 v16bf;
typedef __attribute__((ext_vector_type(8)))  float  v8f;
typedef __attribute__((ext_vector_type(4)))  unsigned int u32x4;
typedef __attribute__((ext_vector_type(4)))  float  f32x4;

struct Frag32 { u32x4 lo, hi; };   // 32 bytes -> one v16bf

__device__ __forceinline__ unsigned int f2bf_rne(float f) {
  unsigned int u = __float_as_uint(f);
  unsigned int r = u + 0x7FFFu + ((u >> 16) & 1u);
  return r >> 16;
}
__device__ __forceinline__ unsigned int pack_bf16x2(float a, float b) {
  return f2bf_rne(a) | (f2bf_rne(b) << 16);
}
__device__ __forceinline__ float sigmoidf_(float x) {
  return 1.0f / (1.0f + __expf(-x));
}

// A-fragment (weights, bf16, row-major [4H,H]), 16x32 (M=units, K):
// lanes 0-15 row m: K [k0..k0+7] and [k0+16..k0+23];
// lanes 16-31 row m: K [k0+8..k0+15] and [k0+24..k0+31].
__device__ __forceinline__ v16bf load_a_frag(const unsigned short* __restrict__ W,
                                             int jjbase, int k0, int lane) {
  const int m  = lane & 15;
  const int kk = k0 + ((lane >> 4) << 3);
  const unsigned short* p = W + (size_t)(jjbase + m) * kH + kk;
  Frag32 f;
  f.lo = *(const u32x4*)(p);        // 8 bf16 = 16B
  f.hi = *(const u32x4*)(p + 16);   // +16 halfs = +32B
  return __builtin_bit_cast(v16bf, f);
}

// B-fragment (activations, bf16, row-major [B,H]), 32x16 (K, N=batch):
// lane n (0-15): column row0+n, K [k0..k0+7] and [k0+16..k0+23];
// lanes 16-31: K [k0+8..k0+15] and [k0+24..k0+31].
__device__ __forceinline__ v16bf load_b_frag(const unsigned short* __restrict__ A,
                                             int row0, int k0, int lane) {
  const int n  = lane & 15;
  const int kk = k0 + ((lane >> 4) << 3);
  const unsigned short* p = A + (size_t)(row0 + n) * kH + kk;
  Frag32 f;
  f.lo = *(const u32x4*)(p);
  f.hi = *(const u32x4*)(p + 16);
  return __builtin_bit_cast(v16bf, f);
}

// One fused LSTM cell step for one layer. No LDS, no barriers: all operands
// are bf16 in global (L2-resident), every wave independent.
// Grid: 32 blocks x 256 threads (8 waves). Block owns 32 hidden units.
// Wave w: unit subtile tj = w&1 (16 units), batch tile bq = w>>1 (16 rows),
// all 4 gates in-register -> acc[4] v8f; fully fused epilogue.
__global__ __launch_bounds__(256)
void lstm_cell_step(const unsigned short* __restrict__ inp_bf,  // [B,H] bf16
                    const unsigned short* __restrict__ hprev_bf,// [B,H] bf16
                    const float* __restrict__ cprev,            // [B,H] f32
                    float* __restrict__ hnew,                   // [B,H] f32
                    unsigned short* __restrict__ hnew_bf,       // [B,H] bf16
                    float* __restrict__ cnew,                   // [B,H] f32
                    const unsigned short* __restrict__ Wxb,     // [4H,H] bf16
                    const unsigned short* __restrict__ Whb,     // [4H,H] bf16
                    const float* __restrict__ bx,               // [4H]
                    const float* __restrict__ bh,               // [4H]
                    float* __restrict__ outp)                   // [B,H] or null
{
  const int tid  = threadIdx.x;
  const int lane = tid & 31;
  const int w    = tid >> 5;
  const int u0   = blockIdx.x * 32 + (w & 1) * 16;  // 16-unit tile base
  const int b0   = (w >> 1) * 16;                   // 16-row batch tile base

  v8f acc[4];
  const v8f vzero = {0.f, 0.f, 0.f, 0.f, 0.f, 0.f, 0.f, 0.f};
#pragma unroll
  for (int g = 0; g < 4; ++g) acc[g] = vzero;

#pragma unroll
  for (int phase = 0; phase < 2; ++phase) {
    const unsigned short* act = phase ? hprev_bf : inp_bf;
    const unsigned short* W   = phase ? Whb : Wxb;
#pragma unroll 4
    for (int k0 = 0; k0 < kH; k0 += 32) {
      const v16bf bfr = load_b_frag(act, b0, k0, lane);
#pragma unroll
      for (int g = 0; g < 4; ++g) {
        const v16bf af = load_a_frag(W, g * kH + u0, k0, lane);
        acc[g] = __builtin_amdgcn_wmma_f32_16x16x32_bf16(
            false, af, false, bfr, (short)0, acc[g], false, false);
      }
    }
  }

  // C/D layout: VGPR e, lanes 0-15 -> M=e ; lanes 16-31 -> M=e+8.
  // j runs contiguously over e => vectorized b128 epilogue.
  const int j0 = u0 + ((lane >> 4) << 3);
  const int b  = b0 + (lane & 15);
  const size_t rowbase = (size_t)b * kH + j0;

  // biases (bx+bh), vectorized
  float bias[4][8];
#pragma unroll
  for (int g = 0; g < 4; ++g) {
    const f32x4 x_lo = *(const f32x4*)(bx + g * kH + j0);
    const f32x4 x_hi = *(const f32x4*)(bx + g * kH + j0 + 4);
    const f32x4 h_lo = *(const f32x4*)(bh + g * kH + j0);
    const f32x4 h_hi = *(const f32x4*)(bh + g * kH + j0 + 4);
#pragma unroll
    for (int e = 0; e < 4; ++e) {
      bias[g][e]     = x_lo[e] + h_lo[e];
      bias[g][e + 4] = x_hi[e] + h_hi[e];
    }
  }

  const f32x4 c_lo = *(const f32x4*)(cprev + rowbase);
  const f32x4 c_hi = *(const f32x4*)(cprev + rowbase + 4);

  float h2v[8], c2v[8];
#pragma unroll
  for (int e = 0; e < 8; ++e) {
    const float xi = acc[0][e] + bias[0][e];
    const float xf = acc[1][e] + bias[1][e];
    const float xg = acc[2][e] + bias[2][e];
    const float xo = acc[3][e] + bias[3][e];
    const float i_ = sigmoidf_(xi);
    const float f_ = sigmoidf_(xf);
    const float g_ = tanhf(xg);
    const float o_ = sigmoidf_(xo);
    const float cp = (e < 4) ? c_lo[e & 3] : c_hi[e & 3];
    const float c2 = f_ * cp + i_ * g_;
    c2v[e] = c2;
    h2v[e] = o_ * tanhf(c2);
  }

  // vectorized stores: c (2xb128), h f32 (2xb128), h bf16 (1xb128), out (2xb128)
  f32x4 h_lo4, h_hi4, cn_lo4, cn_hi4;
#pragma unroll
  for (int e = 0; e < 4; ++e) {
    h_lo4[e]  = h2v[e];     h_hi4[e]  = h2v[e + 4];
    cn_lo4[e] = c2v[e];     cn_hi4[e] = c2v[e + 4];
  }
  *(f32x4*)(cnew + rowbase)     = cn_lo4;
  *(f32x4*)(cnew + rowbase + 4) = cn_hi4;
  *(f32x4*)(hnew + rowbase)     = h_lo4;
  *(f32x4*)(hnew + rowbase + 4) = h_hi4;

  u32x4 hb;
#pragma unroll
  for (int e = 0; e < 4; ++e) hb[e] = pack_bf16x2(h2v[2 * e], h2v[2 * e + 1]);
  *(u32x4*)(hnew_bf + rowbase) = hb;

  if (outp) {
    *(f32x4*)(outp + rowbase)     = h_lo4;
    *(f32x4*)(outp + rowbase + 4) = h_hi4;
  }
}

// fp32 -> bf16 conversion (weights / x / h0), once per kernel_launch
__global__ __launch_bounds__(256)
void f32_to_bf16(const float* __restrict__ src, unsigned short* __restrict__ dst, int n) {
  int i = blockIdx.x * blockDim.x + threadIdx.x;
  const int stride = gridDim.x * blockDim.x;
  for (; i < n; i += stride) dst[i] = (unsigned short)f2bf_rne(src[i]);
}

extern "C" void kernel_launch(void* const* d_in, const int* in_sizes, int n_in,
                              void* d_out, int out_size, void* d_ws, size_t ws_size,
                              hipStream_t stream) {
  const float* x  = (const float*)d_in[0];  // [T,B,H]
  const float* h0 = (const float*)d_in[1];  // [L,B,H]
  const float* c0 = (const float*)d_in[2];  // [L,B,H]
  const float* Wx = (const float*)d_in[3];  // [L,4H,H]
  const float* Wh = (const float*)d_in[4];  // [L,4H,H]
  const float* bx = (const float*)d_in[5];  // [L,4H]
  const float* bh = (const float*)d_in[6];  // [L,4H]
  float* out = (float*)d_out;               // out[T,B,H] ++ hT[L,B,H] ++ cT[L,B,H]

  const size_t nW   = (size_t)kL * kG4H * kH;  // 8.39M per weight tensor
  const size_t nBH  = (size_t)kB * kH;         // 65536
  const size_t nLBH = (size_t)kL * nBH;
  const size_t nX   = (size_t)kT * nBH;        // 16.8M

  // workspace layout
  unsigned short* Wxb = (unsigned short*)d_ws;     // nW bf16
  unsigned short* Whb = Wxb + nW;                  // nW bf16
  unsigned short* xbf = Whb + nW;                  // nX bf16
  unsigned short* hbf = xbf + nX;                  // [2][L,B,H] bf16
  float* hbuf = (float*)(hbf + 2 * nLBH);          // [2][L,B,H] f32
  float* cbuf = hbuf + 2 * nLBH;                   // [2][L,B,H] f32

  // 1) one-time conversions (weights 33MB + x 33MB, L2/HBM streaming)
  f32_to_bf16<<<1024, 256, 0, stream>>>(Wx, Wxb, (int)nW);
  f32_to_bf16<<<1024, 256, 0, stream>>>(Wh, Whb, (int)nW);
  f32_to_bf16<<<2048, 256, 0, stream>>>(x,  xbf, (int)nX);
  f32_to_bf16<<<128,  256, 0, stream>>>(h0, hbf, (int)nLBH);   // parity 0

  // 2) c state ping-pong init (parity 0 = initial)
  hipMemcpyAsync(cbuf, c0, nLBH * sizeof(float), hipMemcpyDeviceToDevice, stream);

  // 3) recurrence: 512 barrier-free fused cell-step kernels
  for (int t = 0; t < kT; ++t) {
    const int rp = t & 1;
    const int wp = rp ^ 1;
    for (int l = 0; l < kL; ++l) {
      const unsigned short* inp =
          (l == 0) ? (xbf + (size_t)t * nBH)
                   : (hbf + (size_t)wp * nLBH + 0 * nBH);  // layer0 out this step
      const unsigned short* hp = hbf + (size_t)rp * nLBH + (size_t)l * nBH;
      const float* cp = cbuf + (size_t)rp * nLBH + (size_t)l * nBH;
      float* hn           = hbuf + (size_t)wp * nLBH + (size_t)l * nBH;
      unsigned short* hnb = hbf  + (size_t)wp * nLBH + (size_t)l * nBH;
      float* cn           = cbuf + (size_t)wp * nLBH + (size_t)l * nBH;
      float* op = (l == kL - 1) ? (out + (size_t)t * nBH) : nullptr;
      lstm_cell_step<<<kH / 32, 256, 0, stream>>>(
          inp, hp, cp, hn, hnb, cn,
          Wxb + (size_t)l * kG4H * kH, Whb + (size_t)l * kG4H * kH,
          bx + (size_t)l * kG4H, bh + (size_t)l * kG4H, op);
    }
  }

  // 4) final states: t=255 wrote parity 0
  hipMemcpyAsync(out + nX, hbuf, nLBH * sizeof(float),
                 hipMemcpyDeviceToDevice, stream);
  hipMemcpyAsync(out + nX + nLBH, cbuf, nLBH * sizeof(float),
                 hipMemcpyDeviceToDevice, stream);
}
